// MultiHeadAttention_69741678952743
// MI455X (gfx1250) — compile-verified, matrix operand order checked by reading, not run on hardware
//
#include <hip/hip_runtime.h>
#include <cstdint>
#include <cstddef>

// ---------------- problem constants ----------------
#define D_MODEL 1024
#define N_HEADS 16
#define D_HEAD  64
#define BATCH   2
#define SEQ     2048
#define MROWS   (BATCH * SEQ)          // 4096 token rows
#define HDH     (N_HEADS * D_HEAD)     // 1024

#define LOG2E 1.4426950408889634f

typedef __attribute__((ext_vector_type(16))) __bf16        v16bf;
typedef __attribute__((ext_vector_type(8)))  float         v8f;
typedef __attribute__((ext_vector_type(4)))  unsigned int  u32x4;
typedef __attribute__((ext_vector_type(8)))  unsigned int  u32x8;

// ---------------- WMMA helpers ----------------
__device__ __forceinline__ v8f wmma_bf16(v16bf a, v16bf b, v8f c) {
  // D(16x16,f32) = A(16x32,bf16) * B(32x16,bf16) + C
  return __builtin_amdgcn_wmma_f32_16x16x32_bf16(
      /*neg_a=*/false, a, /*neg_b=*/false, b,
      /*c_mod=*/(short)0, c, /*reuse_a=*/false, /*reuse_b=*/false);
}

// Load one fragment: elements [kb..kb+7] and [16+kb..16+kb+7] of a
// 32-element K-contiguous row (16-bit A/B VGPR layout: lanes 0-15 kb=0,
// lanes 16-31 kb=8).
__device__ __forceinline__ v16bf load_frag(const __bf16* base, int kb) {
  union { v16bf v; uint4 u[2]; } f;
  f.u[0] = *(const uint4*)(base + kb);
  f.u[1] = *(const uint4*)(base + 16 + kb);
  return f.v;
}

// ---------------- TDM descriptor helpers (CDNA5 ISA ch.8) ----------------
// D# group1 for a 2D bf16 tile with LDS row padding.
// tile0 = elements per row, tile1 = rows, stride0 = global row stride (elems),
// pad_int: bytes-before-pad = 8<<pad_int ; pad_amt: pad DWORDs = pad_amt+1.
__device__ __forceinline__ u32x8 tdm_group1(unsigned tile0, unsigned tile1,
                                            unsigned long long stride0,
                                            unsigned pad_int, unsigned pad_amt) {
  union { unsigned int a[8]; u32x8 v; } g;
  const unsigned td0 = 1u << 30, td1 = 1u << 30;   // huge: never OOB
  g.a[0] = (1u << 16)                 // data_size = 2 bytes
         | (1u << 20)                 // pad_enable
         | (pad_int << 22) | (pad_amt << 25);
  g.a[1] = (td0 & 0xFFFFu) << 16;                          // tensor_dim0[15:0]
  g.a[2] = ((td0 >> 16) & 0xFFFFu) | ((td1 & 0xFFFFu) << 16);
  g.a[3] = ((td1 >> 16) & 0xFFFFu) | (tile0 << 16);        // tile_dim0
  g.a[4] = tile1;                                          // tile_dim1 (dim2=0)
  g.a[5] = (unsigned int)(stride0 & 0xFFFFFFFFull);        // tensor_dim0_stride
  g.a[6] = (unsigned int)((stride0 >> 32) & 0xFFFFull);
  g.a[7] = 0;
  return g.v;
}

// Issue tensor_load_to_lds (EXEC-independent; call from one wave, then
// s_wait_tensorcnt + block barrier before consumers read LDS).
__device__ __forceinline__ void tdm_load(unsigned lds_off, const void* gptr,
                                         u32x8 g1) {
  union { unsigned int a[4]; u32x4 v; } g0;
  const unsigned long long ga = (unsigned long long)(uintptr_t)gptr;
  g0.a[0] = 1u;                                            // count=1, user mode
  g0.a[1] = lds_off;                                       // lds_addr
  g0.a[2] = (unsigned int)ga;                              // global_addr lo
  g0.a[3] = (unsigned int)((ga >> 32) & 0x01FFFFFFull)     // global_addr hi
          | (2u << 30);                                    // type = 2 (image)
  u32x4 z = {0u, 0u, 0u, 0u};
  asm volatile("tensor_load_to_lds %0, %1, %2, %3"
               :: "s"(g0.v), "s"(g1), "s"(z), "s"(z)
               : "memory");
}

// ---------------- f32 -> bf16 convert ----------------
__global__ void cvt_bf16_kernel(const float* __restrict__ in,
                                __bf16* __restrict__ out, int n) {
  int i = blockIdx.x * blockDim.x + threadIdx.x;
  int stride = gridDim.x * blockDim.x;
  for (; i < n; i += stride) out[i] = (__bf16)in[i];
}

// ---------------- bf16 GEMM: out[M,N] = A[M,K] @ B[K,N] + bias ----------------
// 128x64 tile / block, 256 threads = 8 waves (4x2), each wave 32x32
// (2 A-frags x 2 B-frags -> 4 WMMAs per 32-K step).
// A tile arrives via the Tensor Data Mover (TDM pads rows 64B+16B -> stride 40).
__global__ __launch_bounds__(256) void gemm_bf16_kernel(
    const __bf16* __restrict__ A, const __bf16* __restrict__ Bm,
    const float* __restrict__ bias, float* __restrict__ outF,
    __bf16* __restrict__ outB, int M, int N, int K) {
  constexpr int AS = 40;                    // padded row stride (conflict-free)
  __shared__ __bf16 As[128 * AS];           // A tile [m][k] 128x32 (TDM-filled)
  __shared__ __bf16 Bt[64 * AS];            // B tile transposed [n][k] 64x32

  const int tid  = threadIdx.x;
  const int lane = tid & 31;
  const int w    = tid >> 5;
  const int wm   = w >> 1;                  // 0..3 -> 32-row slice
  const int wn   = w & 1;                   // 0..1 -> 32-col slice
  const int m0   = blockIdx.y * 128;
  const int n0   = blockIdx.x * 64;
  const int fm   = lane & 15;
  const int hi   = lane >> 4;
  const int kb   = hi * 8;

  const int brow = tid >> 3, bcol = (tid & 7) * 8;  // B-tile cooperative load

  // TDM: 32-elem rows (64B) then 16B pad -> pad_int=3 (8<<3=64B), pad_amt=3 (4 DW)
  const u32x8 g1A = tdm_group1(/*tile0=*/32, /*tile1=*/128,
                               /*stride0=*/(unsigned long long)K,
                               /*pad_int=*/3, /*pad_amt=*/3);
  const unsigned ldsA = (unsigned)(uintptr_t)&As[0];

  v8f acc00 = {}, acc01 = {}, acc10 = {}, acc11 = {};

  for (int kt = 0; kt < K; kt += 32) {
    __syncthreads();
    if (w == 0)   // async DMA of the 128x32 A tile into LDS
      tdm_load(ldsA, A + (size_t)m0 * K + kt, g1A);
    // B tile: load row-major, store transposed into Bt[n][k]
    union { uint4 u; __bf16 h[8]; } tb;
    tb.u = *(const uint4*)&Bm[(size_t)(kt + brow) * N + n0 + bcol];
#pragma unroll
    for (int e = 0; e < 8; ++e) Bt[(bcol + e) * AS + brow] = tb.h[e];
    if (w == 0) __builtin_amdgcn_s_wait_tensorcnt(0);
    __syncthreads();

    v16bf af0 = load_frag(&As[(wm * 32 + fm) * AS], kb);
    v16bf af1 = load_frag(&As[(wm * 32 + 16 + fm) * AS], kb);
    v16bf bf0 = load_frag(&Bt[(wn * 32 + fm) * AS], kb);
    v16bf bf1 = load_frag(&Bt[(wn * 32 + 16 + fm) * AS], kb);
    acc00 = wmma_bf16(af0, bf0, acc00);
    acc01 = wmma_bf16(af0, bf1, acc01);
    acc10 = wmma_bf16(af1, bf0, acc10);
    acc11 = wmma_bf16(af1, bf1, acc11);
  }

  // Epilogue: C/D layout -> lane holds (n = lane&15, m = r + 8*(lane>>4))
#pragma unroll
  for (int r = 0; r < 8; ++r) {
    const int gn = n0 + wn * 32 + fm;
    const int mA = m0 + wm * 32 + r + 8 * hi;
    const int mB = mA + 16;
    const float v00 = acc00[r] + bias[gn];
    const float v01 = acc01[r] + bias[gn + 16];
    const float v10 = acc10[r] + bias[gn];
    const float v11 = acc11[r] + bias[gn + 16];
    const size_t oA = (size_t)mA * N + gn;
    const size_t oB = (size_t)mB * N + gn;
    if (outF) {
      outF[oA] = v00; outF[oA + 16] = v01;
      outF[oB] = v10; outF[oB + 16] = v11;
    }
    if (outB) {
      outB[oA] = (__bf16)v00; outB[oA + 16] = (__bf16)v01;
      outB[oB] = (__bf16)v10; outB[oB + 16] = (__bf16)v11;
    }
  }
}

// ---------------- flash attention ----------------
// grid = (SEQ/64 q-blocks, BATCH*N_HEADS); block = 128 threads = 4 waves.
// Each wave owns 16 query rows; loops over 32 key blocks of 64 keys.
// K tile arrives via TDM (rows 128B + 16B pad -> stride 72); V transposed manually.
__global__ __launch_bounds__(128) void attn_kernel(
    const __bf16* __restrict__ q,    // [MROWS, HDH] bf16
    const __bf16* __restrict__ kv,   // [MROWS, 2*HDH] bf16 (k | v)
    const int* __restrict__ mask,    // [BATCH, SEQ]
    __bf16* __restrict__ attO) {     // [MROWS, HDH] bf16
  constexpr int KS = 72;                    // padded stride for 64-wide rows
  __shared__ __bf16 Ks[64 * KS];            // K tile  [key][d] (TDM-filled)
  __shared__ __bf16 Vt[64 * KS];            // V tile  [d][key] (transposed)
  __shared__ __bf16 Pst[4 * 16 * KS];       // per-wave P staging [row][key]
  __shared__ float  maskadd[64];

  const int tid  = threadIdx.x;
  const int lane = tid & 31;
  const int w    = tid >> 5;                // wave 0..3
  const int fm   = lane & 15;
  const int hi   = lane >> 4;
  const int kb   = hi * 8;
  const int qblk = blockIdx.x;
  const int b    = blockIdx.y >> 4;
  const int hh   = blockIdx.y & 15;
  const float scale = 0.125f;               // 1/sqrt(64)

  // TDM: 64-elem rows (128B) then 16B pad -> pad_int=4 (8<<4=128B), pad_amt=3
  const u32x8 g1K = tdm_group1(/*tile0=*/64, /*tile1=*/64,
                               /*stride0=*/2ull * HDH,
                               /*pad_int=*/4, /*pad_amt=*/3);
  const unsigned ldsK = (unsigned)(uintptr_t)&Ks[0];

  // Q fragments stay in registers (A layout)
  const int qrow = qblk * 64 + w * 16 + fm;
  const __bf16* qbase = q + (size_t)(b * SEQ + qrow) * HDH + hh * D_HEAD;
  v16bf qf[2];
  qf[0] = load_frag(qbase, kb);
  qf[1] = load_frag(qbase + 32, kb);

  v8f   oacc[4] = {{}, {}, {}, {}};
  float mrow[8], lrow[8];
#pragma unroll
  for (int r = 0; r < 8; ++r) { mrow[r] = -1e30f; lrow[r] = 0.f; }

  __bf16* Pw = &Pst[w * 16 * KS];

  for (int kblk = 0; kblk < SEQ / 64; ++kblk) {
    __syncthreads();
    const __bf16* kvbase =
        kv + (size_t)(b * SEQ + kblk * 64) * (2 * HDH) + hh * D_HEAD;
    if (w == 0)   // async DMA of the 64x64 K tile into LDS
      tdm_load(ldsK, kvbase, g1K);
    // V tile: manual transposed stage [d][key]
#pragma unroll
    for (int i = 0; i < 4; ++i) {
      const int idx = tid + i * 128;        // 0..511 -> 64 rows x 8 chunks
      const int kr  = idx >> 3;
      const int c   = (idx & 7) * 8;
      union { uint4 u; __bf16 h[8]; } tv;
      tv.u = *(const uint4*)(kvbase + (size_t)kr * (2 * HDH) + HDH + c);
#pragma unroll
      for (int e = 0; e < 8; ++e) Vt[(c + e) * KS + kr] = tv.h[e];
    }
    if (tid < 64)
      maskadd[tid] = mask[b * SEQ + kblk * 64 + tid] ? -1e9f : 0.0f;
    if (w == 0) __builtin_amdgcn_s_wait_tensorcnt(0);
    __syncthreads();

    // scores S = Q K^T * scale + mask   (4 n-tiles x 2 k-steps)
    v8f sacc[4] = {{}, {}, {}, {}};
#pragma unroll
    for (int ks = 0; ks < 2; ++ks) {
#pragma unroll
      for (int nt = 0; nt < 4; ++nt) {
        v16bf bf = load_frag(&Ks[(nt * 16 + fm) * KS + ks * 32], kb);
        sacc[nt] = wmma_bf16(qf[ks], bf, sacc[nt]);
      }
    }
    float sv[4][8], pv[4][8];
#pragma unroll
    for (int nt = 0; nt < 4; ++nt) {
      const float ma = maskadd[nt * 16 + fm];
#pragma unroll
      for (int r = 0; r < 8; ++r) sv[nt][r] = sacc[nt][r] * scale + ma;
    }

    // online softmax per row (rows live in 16-lane groups; butterfly over n)
#pragma unroll
    for (int r = 0; r < 8; ++r) {
      float t = fmaxf(fmaxf(sv[0][r], sv[1][r]), fmaxf(sv[2][r], sv[3][r]));
      t = fmaxf(t, __shfl_xor(t, 1));
      t = fmaxf(t, __shfl_xor(t, 2));
      t = fmaxf(t, __shfl_xor(t, 4));
      t = fmaxf(t, __shfl_xor(t, 8));
      const float mn   = fmaxf(mrow[r], t);
      const float corr = exp2f((mrow[r] - mn) * LOG2E);
      mrow[r] = mn;
      float s = 0.f;
#pragma unroll
      for (int nt = 0; nt < 4; ++nt) {
        const float p = exp2f((sv[nt][r] - mn) * LOG2E);
        pv[nt][r] = p;
        s += p;
      }
      s += __shfl_xor(s, 1);
      s += __shfl_xor(s, 2);
      s += __shfl_xor(s, 4);
      s += __shfl_xor(s, 8);
      lrow[r] = lrow[r] * corr + s;
#pragma unroll
      for (int nd = 0; nd < 4; ++nd) oacc[nd][r] *= corr;
    }

    // stage P (bf16) in A layout [row][key]
#pragma unroll
    for (int nt = 0; nt < 4; ++nt)
#pragma unroll
      for (int r = 0; r < 8; ++r)
        Pw[(r + 8 * hi) * KS + nt * 16 + fm] = (__bf16)pv[nt][r];
    __syncthreads();

    // O += P @ V    (4 d-tiles x 2 k-steps over keys)
#pragma unroll
    for (int ks = 0; ks < 2; ++ks) {
      v16bf pf = load_frag(&Pw[fm * KS + ks * 32], kb);
#pragma unroll
      for (int nd = 0; nd < 4; ++nd) {
        v16bf vf = load_frag(&Vt[(nd * 16 + fm) * KS + ks * 32], kb);
        oacc[nd] = wmma_bf16(pf, vf, oacc[nd]);
      }
    }
  }

  // normalize and write att (bf16 for the Wo GEMM)
#pragma unroll
  for (int nd = 0; nd < 4; ++nd) {
#pragma unroll
    for (int r = 0; r < 8; ++r) {
      const float val = oacc[nd][r] / lrow[r];
      const int row = qblk * 64 + w * 16 + r + 8 * hi;
      const int d   = nd * 16 + fm;
      attO[(size_t)(b * SEQ + row) * HDH + hh * D_HEAD + d] = (__bf16)val;
    }
  }
}

// ---------------- fused residual + GLU + LayerNorm ----------------
__global__ __launch_bounds__(256) void fuse_ln_kernel(
    const float* __restrict__ h, const float* __restrict__ att,
    const float* __restrict__ s1, const float* __restrict__ s2,
    const float* __restrict__ g, const float* __restrict__ bb,
    float* __restrict__ out) {
  __shared__ float r1[256], r2[256];
  const int row = blockIdx.x;
  const int tid = threadIdx.x;
  const size_t base = (size_t)row * D_MODEL;

  float xv[4];
  float sum = 0.f, sq = 0.f;
#pragma unroll
  for (int k = 0; k < 4; ++k) {
    const int i = tid + k * 256;
    const float a  = s1[base + i];
    const float gl = (1.f / (1.f + expf(-a))) * s2[base + i];
    const float x  = h[base + i] + att[base + i] + gl;
    xv[k] = x;
    sum += x;
    sq  += x * x;
  }
  r1[tid] = sum;
  r2[tid] = sq;
  __syncthreads();
  for (int st = 128; st > 0; st >>= 1) {
    if (tid < st) { r1[tid] += r1[tid + st]; r2[tid] += r2[tid + st]; }
    __syncthreads();
  }
  const float mean = r1[0] * (1.f / D_MODEL);
  const float var  = r2[0] * (1.f / D_MODEL) - mean * mean;
  const float rstd = rsqrtf(var + 1e-6f);
#pragma unroll
  for (int k = 0; k < 4; ++k) {
    const int i = tid + k * 256;
    out[base + i] = (xv[k] - mean) * rstd * g[i] + bb[i];
  }
}

// ---------------- launch ----------------
extern "C" void kernel_launch(void* const* d_in, const int* in_sizes, int n_in,
                              void* d_out, int out_size, void* d_ws, size_t ws_size,
                              hipStream_t stream) {
  const float* h    = (const float*)d_in[0];
  const int*   mask = (const int*)d_in[1];
  const float* Wq   = (const float*)d_in[2];
  const float* bq   = (const float*)d_in[3];
  const float* Wkv  = (const float*)d_in[4];
  const float* bkv  = (const float*)d_in[5];
  const float* Wo   = (const float*)d_in[6];
  const float* bo   = (const float*)d_in[7];
  const float* W1   = (const float*)d_in[8];
  const float* b1   = (const float*)d_in[9];
  const float* W2   = (const float*)d_in[10];
  const float* b2   = (const float*)d_in[11];
  const float* lng  = (const float*)d_in[12];
  const float* lnb  = (const float*)d_in[13];
  float* out = (float*)d_out;

  // workspace carve-up
  char* ws = (char*)d_ws;
  size_t off = 0;
  auto take = [&](size_t bytes) { char* p = ws + off; off += bytes; return p; };
  __bf16* h_bf   = (__bf16*)take((size_t)MROWS * D_MODEL * 2);      //  8 MB
  __bf16* q_bf   = (__bf16*)take((size_t)MROWS * HDH * 2);          //  8 MB
  __bf16* kv_bf  = (__bf16*)take((size_t)MROWS * 2 * HDH * 2);      // 16 MB
  __bf16* att_bf = (__bf16*)take((size_t)MROWS * HDH * 2);          //  8 MB
  __bf16* wq_bf  = (__bf16*)take((size_t)D_MODEL * HDH * 2);        //  2 MB
  __bf16* wkv_bf = (__bf16*)take((size_t)D_MODEL * 2 * HDH * 2);    //  4 MB
  __bf16* wo_bf  = (__bf16*)take((size_t)HDH * D_MODEL * 2);        //  2 MB
  __bf16* w1_bf  = (__bf16*)take((size_t)D_MODEL * D_MODEL * 2);    //  2 MB
  __bf16* w2_bf  = (__bf16*)take((size_t)D_MODEL * D_MODEL * 2);    //  2 MB
  float*  atto_f = (float*)take((size_t)MROWS * D_MODEL * 4);       // 16 MB
  float*  s1_f   = (float*)take((size_t)MROWS * D_MODEL * 4);       // 16 MB
  float*  s2_f   = (float*)take((size_t)MROWS * D_MODEL * 4);       // 16 MB

  // 1) convert activations + weights to bf16
  cvt_bf16_kernel<<<2048, 256, 0, stream>>>(h,   h_bf,   MROWS * D_MODEL);
  cvt_bf16_kernel<<<1024, 256, 0, stream>>>(Wq,  wq_bf,  D_MODEL * HDH);
  cvt_bf16_kernel<<<2048, 256, 0, stream>>>(Wkv, wkv_bf, D_MODEL * 2 * HDH);
  cvt_bf16_kernel<<<1024, 256, 0, stream>>>(Wo,  wo_bf,  HDH * D_MODEL);
  cvt_bf16_kernel<<<1024, 256, 0, stream>>>(W1,  w1_bf,  D_MODEL * D_MODEL);
  cvt_bf16_kernel<<<1024, 256, 0, stream>>>(W2,  w2_bf,  D_MODEL * D_MODEL);

  // 2) projections (bf16 WMMA GEMMs, 128x64 tiles)
  gemm_bf16_kernel<<<dim3(HDH / 64, MROWS / 128), 256, 0, stream>>>(
      h_bf, wq_bf, bq, nullptr, q_bf, MROWS, HDH, D_MODEL);
  gemm_bf16_kernel<<<dim3(2 * HDH / 64, MROWS / 128), 256, 0, stream>>>(
      h_bf, wkv_bf, bkv, nullptr, kv_bf, MROWS, 2 * HDH, D_MODEL);

  // 3) flash attention (TDM K tiles + WMMA)
  attn_kernel<<<dim3(SEQ / 64, BATCH * N_HEADS), 128, 0, stream>>>(
      q_bf, kv_bf, mask, att_bf);

  // 4) output projection + MLP branches
  gemm_bf16_kernel<<<dim3(D_MODEL / 64, MROWS / 128), 256, 0, stream>>>(
      att_bf, wo_bf, bo, atto_f, nullptr, MROWS, D_MODEL, HDH);
  gemm_bf16_kernel<<<dim3(D_MODEL / 64, MROWS / 128), 256, 0, stream>>>(
      h_bf, w1_bf, b1, s1_f, nullptr, MROWS, D_MODEL, D_MODEL);
  gemm_bf16_kernel<<<dim3(D_MODEL / 64, MROWS / 128), 256, 0, stream>>>(
      h_bf, w2_bf, b2, s2_f, nullptr, MROWS, D_MODEL, D_MODEL);

  // 5) fused residual + GLU + LayerNorm
  fuse_ln_kernel<<<MROWS, 256, 0, stream>>>(h, atto_f, s1_f, s2_f, lng, lnb, out);

  (void)in_sizes; (void)n_in; (void)out_size; (void)ws_size;
}
